// WinnerTakeAllSNN_4913442587077
// MI455X (gfx1250) — matrix operand order repaired and checked by reading, not compiled, not run on hardware
//
#include <hip/hip_runtime.h>

#define HIDDEN_N 800
#define TSTEPS   50
#define BATCH_N  1024
#define INDIM    784
#define OUTDIM   10
#define KPAD     800      // K padded to 25 * 32
#define N3PAD    32       // layer-3 output cols padded to one 32-wide wave tile
#define BETA_C   0.95f
#define THRESH_C 1.0f

typedef __attribute__((ext_vector_type(16))) _Float16 v16h;
typedef __attribute__((ext_vector_type(8)))  _Float16 v8h;
typedef __attribute__((ext_vector_type(8)))  float    v8f;

// ---------------------------------------------------------------------------
// Branch-free f16 fragment loads (CDNA5 16x16x32 layouts, CDNA5 ISA §7.12.2).
// A (16x32 MxK): lane m<16 holds K[0:8]+K[16:24]; lane m+16 holds K[8:16]+K[24:32].
//   Per-lane base = row(m) + (lane>>4)*8; chunks at +0 and +16 halfs.
// B (32x16 KxN) with B[k][n]=W[n][k]: lane n<16 holds W[n][k0:k0+8]+W[n][k0+16:k0+24],
//   lane n+16 holds W[n][k0+8:k0+16]+W[n][k0+24:k0+32].  Per-lane base = row(n) + (lane>>4)*8.
// ---------------------------------------------------------------------------
__device__ __forceinline__ v16h ldA(const _Float16* __restrict__ p) {
  v8h lo = *(const v8h*)(p);
  v8h hi = *(const v8h*)(p + 16);
  return __builtin_shufflevector(lo, hi, 0,1,2,3,4,5,6,7,8,9,10,11,12,13,14,15);
}
__device__ __forceinline__ v16h ldB(const _Float16* __restrict__ p) {
  v8h lo = *(const v8h*)(p);
  v8h hi = *(const v8h*)(p + 16);
  return __builtin_shufflevector(lo, hi, 0,1,2,3,4,5,6,7,8,9,10,11,12,13,14,15);
}

#define WMMA_F16(a, b, c) \
  __builtin_amdgcn_wmma_f32_16x16x32_f16(false, (a), false, (b), (short)0, (c), false, false)

// C/D 16x16 f32 layout: lane L<16 -> col L, row r = VGPR r; lanes 16-31 -> row r+8.
__device__ __forceinline__ void store_tile(float* __restrict__ C, int ldc,
                                           int rowbase, int colbase, v8f acc,
                                           const float* __restrict__ bias, int lane) {
  const int col  = colbase + (lane & 15);
  const int rowb = rowbase + (lane >> 4) * 8;
  const float bv = bias[col];
#pragma unroll
  for (int r = 0; r < 8; ++r)
    C[(long)(rowb + r) * ldc + col] = acc[r] + bv;
}

// C[M x Ncols] = Ah[M x K] @ Bh[Ncols x K]^T + bias, all f16 in / f32 out.
// One wave per 64x32 tile: 4x2 WMMA tiles, 8 v_wmma per k-step, branch-free.
__global__ __launch_bounds__(256) void gemm_f16_wmma(
    const _Float16* __restrict__ Ah, const _Float16* __restrict__ Bh,
    const float* __restrict__ bias, float* __restrict__ C,
    int K, int lda, int ldb, int ldc, int mtiles, int ntiles) {
  const int wave = blockIdx.x * (blockDim.x >> 5) + (threadIdx.x >> 5);
  const int lane = threadIdx.x & 31;
  const int mt = wave % mtiles;
  const int nt = wave / mtiles;
  if (nt >= ntiles) return;
  const int row0 = mt * 64;
  const int col0 = nt * 32;
  const int l15 = lane & 15;
  const int hh  = lane >> 4;

  const _Float16* a0p = Ah + (long)(row0 +      l15) * lda + hh * 8;
  const _Float16* a1p = Ah + (long)(row0 + 16 + l15) * lda + hh * 8;
  const _Float16* a2p = Ah + (long)(row0 + 32 + l15) * lda + hh * 8;
  const _Float16* a3p = Ah + (long)(row0 + 48 + l15) * lda + hh * 8;
  const _Float16* b0p = Bh + (long)(col0 +      l15) * ldb + hh * 8;
  const _Float16* b1p = Bh + (long)(col0 + 16 + l15) * ldb + hh * 8;

  v8f acc[4][2] = {};
  for (int k0 = 0; k0 < K; k0 += 32) {
    __builtin_prefetch(a0p + k0 + 32, 0, 1);   // global_prefetch_b8, next k-band
    const v16h b0 = ldB(b0p + k0);
    const v16h b1 = ldB(b1p + k0);
    const v16h a0 = ldA(a0p + k0);
    const v16h a1 = ldA(a1p + k0);
    const v16h a2 = ldA(a2p + k0);
    const v16h a3 = ldA(a3p + k0);
    acc[0][0] = WMMA_F16(a0, b0, acc[0][0]);
    acc[0][1] = WMMA_F16(a0, b1, acc[0][1]);
    acc[1][0] = WMMA_F16(a1, b0, acc[1][0]);
    acc[1][1] = WMMA_F16(a1, b1, acc[1][1]);
    acc[2][0] = WMMA_F16(a2, b0, acc[2][0]);
    acc[2][1] = WMMA_F16(a2, b1, acc[2][1]);
    acc[3][0] = WMMA_F16(a3, b0, acc[3][0]);
    acc[3][1] = WMMA_F16(a3, b1, acc[3][1]);
  }
#pragma unroll
  for (int i = 0; i < 4; ++i) {
    store_tile(C, ldc, row0 + i * 16, col0,      acc[i][0], bias, lane);
    store_tile(C, ldc, row0 + i * 16, col0 + 16, acc[i][1], bias, lane);
  }
}

// f32 -> f16 with zero padding (rows and cols).  blockIdx.y = dest row.
__global__ void cvt_pad_f16(const float* __restrict__ src, _Float16* __restrict__ dst,
                            int srcRows, int srcCols, int dstCols) {
  const int c = blockIdx.x * blockDim.x + threadIdx.x;
  const int r = blockIdx.y;
  if (c >= dstCols) return;
  const float v = (r < srcRows && c < srcCols) ? src[(long)r * srcCols + c] : 0.0f;
  dst[(long)r * dstCols + c] = (_Float16)v;
}

__global__ void pad_bias_f32(const float* __restrict__ src, float* __restrict__ dst,
                             int n, int np) {
  const int i = blockIdx.x * blockDim.x + threadIdx.x;
  if (i < np) dst[i] = (i < n) ? src[i] : 0.0f;
}

// ---------------------------------------------------------------------------
// Winner-take-all membrane scan: one 800-thread (25-wave) block per batch row,
// membrane in a register.  Exact k-th largest of 800 by 32-step bisection over
// order-preserving uint32 keys; counts via wave32 ballot + LDS reduction.
// Emits spikes both as f32 (output) and f16 (next layer's GEMM operand).
// ---------------------------------------------------------------------------
__global__ __launch_bounds__(1024) void scan_wta_kernel(
    const float* __restrict__ Z, float* __restrict__ spk,
    _Float16* __restrict__ spk16, const int* __restrict__ kp,
    int T, int B, int H) {
  __shared__ int wcount[32];
  const int b    = blockIdx.x;
  const int h    = threadIdx.x;
  const int wid  = h >> 5;
  const int lane = h & 31;
  const int nw   = (H + 31) >> 5;
  const int k    = *kp;

  float m = 0.0f;
  for (int t = 0; t < T; ++t) {
    const long idx = ((long)t * B + b) * H + h;
    m = BETA_C * m + Z[idx];
    const unsigned u   = __float_as_uint(m);
    const unsigned key = (u & 0x80000000u) ? ~u : (u | 0x80000000u);
    unsigned cur = 0u;
    for (int bit = 31; bit >= 0; --bit) {
      const unsigned cand = cur | (1u << bit);
      const unsigned bal  = __builtin_amdgcn_ballot_w32(key >= cand);
      if (lane == 0) wcount[wid] = __builtin_popcount(bal);
      __syncthreads();
      int tot = 0;
      for (int w = 0; w < nw; ++w) tot += wcount[w];
      if (tot >= k) cur = cand;   // uniform across the block
      __syncthreads();
    }
    const float mask = (key >= cur) ? 1.0f : 0.0f;
    const float s    = (m > THRESH_C) ? mask : 0.0f;
    spk[idx]   = s;
    spk16[idx] = (_Float16)s;
    m *= (1.0f - s);
  }
}

// Layer-3 scan: no top-k coupling; one thread per (batch, class).
__global__ void scan_out_kernel(const float* __restrict__ Z3,
                                float* __restrict__ spk3,
                                float* __restrict__ ssum,
                                int T, int B, int ldz) {
  const int idx = blockIdx.x * blockDim.x + threadIdx.x;
  if (idx >= B * OUTDIM) return;
  const int b = idx / OUTDIM;
  const int j = idx % OUTDIM;
  float m = 0.0f, acc = 0.0f;
  for (int t = 0; t < T; ++t) {
    m = BETA_C * m + Z3[((long)t * B + b) * ldz + j];
    const float s = (m > THRESH_C) ? 1.0f : 0.0f;
    spk3[((long)t * B + b) * OUTDIM + j] = s;
    acc += s;
    m *= (1.0f - s);
  }
  ssum[b * OUTDIM + j] = acc;
}

extern "C" void kernel_launch(void* const* d_in, const int* in_sizes, int n_in,
                              void* d_out, int out_size, void* d_ws, size_t ws_size,
                              hipStream_t stream) {
  const float* x  = (const float*)d_in[0];
  const float* W1 = (const float*)d_in[1];
  const float* b1 = (const float*)d_in[2];
  const float* W2 = (const float*)d_in[3];
  const float* b2 = (const float*)d_in[4];
  const float* W3 = (const float*)d_in[5];
  const float* b3 = (const float*)d_in[6];
  const int*   kp = (const int*)d_in[7];
  float* out = (float*)d_out;

  const int M = TSTEPS * BATCH_N;   // 51200 rows: all timesteps batched

  // Workspace layout (bytes, all 16B-aligned):
  //   Z   : M*800*4      = 163,840,000   (reused for Z1, Z2, Z3)
  //   Ah  : M*800*2      =  81,920,000   (x_h, then spk1_h, then spk2_h)
  //   W1h : 800*800*2    =   1,280,000
  //   W2h : 800*800*2    =   1,280,000
  //   W3h : 32*800*2     =      51,200
  //   b3p : 32*4         =         128
  char* ws = (char*)d_ws;
  float*    Z   = (float*)ws;
  _Float16* Ah  = (_Float16*)(ws + 163840000L);
  _Float16* W1h = (_Float16*)(ws + 245760000L);
  _Float16* W2h = (_Float16*)(ws + 247040000L);
  _Float16* W3h = (_Float16*)(ws + 248320000L);
  float*    b3p = (float*)   (ws + 248371200L);

  // Output layout: [sum(B,10) | spk1(T,B,800) | spk2(T,B,800) | spk3(T,B,10)]
  const long o_s1 = (long)BATCH_N * OUTDIM;
  const long o_s2 = o_s1 + (long)M * HIDDEN_N;
  const long o_s3 = o_s2 + (long)M * HIDDEN_N;
  float* spk1 = out + o_s1;
  float* spk2 = out + o_s2;
  float* spk3 = out + o_s3;

  // --- one-time f16 staging (pads K 784->800, W3 rows 10->32) ---
  {
    dim3 blk(256);
    hipLaunchKernelGGL(cvt_pad_f16, dim3((KPAD + 255) / 256, M),        blk, 0, stream,
                       x,  Ah,  M,        INDIM,    KPAD);
    hipLaunchKernelGGL(cvt_pad_f16, dim3((KPAD + 255) / 256, HIDDEN_N), blk, 0, stream,
                       W1, W1h, HIDDEN_N, INDIM,    KPAD);
    hipLaunchKernelGGL(cvt_pad_f16, dim3((KPAD + 255) / 256, HIDDEN_N), blk, 0, stream,
                       W2, W2h, HIDDEN_N, HIDDEN_N, KPAD);
    hipLaunchKernelGGL(cvt_pad_f16, dim3((KPAD + 255) / 256, N3PAD),    blk, 0, stream,
                       W3, W3h, OUTDIM,   HIDDEN_N, KPAD);
    hipLaunchKernelGGL(pad_bias_f32, dim3(1), dim3(64), 0, stream, b3, b3p, OUTDIM, N3PAD);
  }

  auto launch_gemm = [&](const _Float16* Ap, const _Float16* Bp, const float* biasp,
                         float* Cp, int ldc, int Ncols) {
    const int mtiles = M / 64;               // 800
    const int ntiles = Ncols / 32;           // 25 or 1
    const long waves = (long)mtiles * ntiles;
    const int wavesPerBlock = 256 / 32;
    const int blocks = (int)((waves + wavesPerBlock - 1) / wavesPerBlock);
    hipLaunchKernelGGL(gemm_f16_wmma, dim3(blocks), dim3(256), 0, stream,
                       Ap, Bp, biasp, Cp, KPAD, KPAD, KPAD, ldc, mtiles, ntiles);
  };

  // Layer 1: Z1 = X @ W1^T + b1
  launch_gemm(Ah, W1h, b1, Z, HIDDEN_N, HIDDEN_N);
  hipLaunchKernelGGL(scan_wta_kernel, dim3(BATCH_N), dim3(HIDDEN_N), 0, stream,
                     Z, spk1, Ah, kp, TSTEPS, BATCH_N, HIDDEN_N);   // Ah := spk1_h

  // Layer 2: Z2 = spk1 @ W2^T + b2
  launch_gemm(Ah, W2h, b2, Z, HIDDEN_N, HIDDEN_N);
  hipLaunchKernelGGL(scan_wta_kernel, dim3(BATCH_N), dim3(HIDDEN_N), 0, stream,
                     Z, spk2, Ah, kp, TSTEPS, BATCH_N, HIDDEN_N);   // Ah := spk2_h

  // Layer 3: Z3 = spk2 @ W3^T + b3  (N padded to 32, ldc = 32)
  launch_gemm(Ah, W3h, b3p, Z, N3PAD, N3PAD);
  hipLaunchKernelGGL(scan_out_kernel, dim3((BATCH_N * OUTDIM + 255) / 256), dim3(256), 0, stream,
                     Z, spk3, out, TSTEPS, BATCH_N, N3PAD);
}
